// GPNNCell_2869038154356
// MI455X (gfx1250) — compile-verified, matrix operand order checked by reading, not run on hardware
//
#include <hip/hip_runtime.h>
#include <hip/hip_bf16.h>
#include <cstdint>
#include <cstddef>

// ---------------------------------------------------------------------------
// Types for gfx1250 WMMA (wave32): v_wmma_f32_16x16x32_bf16
// ---------------------------------------------------------------------------
typedef __bf16 bf16;
typedef __attribute__((ext_vector_type(16))) bf16  v16bf;
typedef __attribute__((ext_vector_type(8)))  bf16  v8bf;
typedef __attribute__((ext_vector_type(8)))  float v8f;

namespace cfg {
constexpr int D  = 768;
constexpr int D3 = 2304;
constexpr int D2 = 1536;
constexpr int FF = 3072;
constexpr int NH = 12;
constexpr int HD = 64;
constexpr int B  = 64;
constexpr int F  = 16;
constexpr int N  = 9;
constexpr int TN = 18;            // 2*N
constexpr int R  = B * F * TN;    // 18432 rows through the big GEMMs
constexpr int S  = B * TN;        // 1152 sequences, T = F = 16 tokens
}

// ---------------------------------------------------------------------------
// Device helpers
// ---------------------------------------------------------------------------
__device__ __forceinline__ float gelu_exact(float x) {
    return 0.5f * x * (1.0f + erff(x * 0.70710678118654752f));
}

__device__ __forceinline__ float sigmoidf(float x) {
    return 1.0f / (1.0f + expf(-x));
}

// Block-wide reduction of two running sums (256 threads = 8 waves of 32).
__device__ __forceinline__ void blockReduce2(float& s1, float& s2) {
    #pragma unroll
    for (int off = 16; off > 0; off >>= 1) {
        s1 += __shfl_down(s1, off, 32);
        s2 += __shfl_down(s2, off, 32);
    }
    __shared__ float a1[8], a2[8];
    const int lane = threadIdx.x & 31;
    const int w    = threadIdx.x >> 5;
    if (lane == 0) { a1[w] = s1; a2[w] = s2; }
    __syncthreads();
    if (w == 0) {
        s1 = (lane < 8) ? a1[lane] : 0.0f;
        s2 = (lane < 8) ? a2[lane] : 0.0f;
        #pragma unroll
        for (int off = 4; off > 0; off >>= 1) {
            s1 += __shfl_down(s1, off, 32);
            s2 += __shfl_down(s2, off, 32);
        }
        if (lane == 0) { a1[0] = s1; a2[0] = s2; }
    }
    __syncthreads();
    s1 = a1[0];
    s2 = a2[0];
}

// Assemble a v16bf WMMA fragment from two contiguous 16-byte chunks.
// 16-bit A/B layout (ISA 7.12.2): lane L holds row (L&15); halves 0..7 are
// K = koff..koff+7 and halves 8..15 are K = 16+koff..16+koff+7, koff=(L>=16)*8.
__device__ __forceinline__ v16bf load_frag(const bf16* __restrict__ row, int k0, int koff) {
    const v8bf lo = *reinterpret_cast<const v8bf*>(row + k0 + koff);
    const v8bf hi = *reinterpret_cast<const v8bf*>(row + k0 + 16 + koff);
    v16bf f;
    #pragma unroll
    for (int i = 0; i < 8; ++i) { f[i] = lo[i]; f[i + 8] = hi[i]; }
    return f;
}

// ---------------------------------------------------------------------------
// Weight pack: W[K][N] fp32 row-major  ->  Wt[N][K] bf16 (WMMA B-fragment order)
// ---------------------------------------------------------------------------
__global__ void pack_wt_kernel(const float* __restrict__ W, bf16* __restrict__ out,
                               int K, int N) {
    size_t idx = (size_t)blockIdx.x * 256 + threadIdx.x;
    if (idx >= (size_t)K * N) return;
    int n = (int)(idx / K);
    int k = (int)(idx % K);
    out[idx] = (bf16)W[(size_t)k * N + n];
}

// ---------------------------------------------------------------------------
// Generic bf16 WMMA GEMM:  out = act(A[M,K] @ Bt[N,K]^T + bias)
// 1 wave = one 16x64 tile (A fragment reused 4x). 256 threads = 8 waves.
// ---------------------------------------------------------------------------
__global__ void gemm_bf16_kernel(const bf16* __restrict__ A, const bf16* __restrict__ Bt,
                                 const float* __restrict__ bias,
                                 float* __restrict__ out32, bf16* __restrict__ out16,
                                 int M, int Nc, int K, int doGelu) {
    const int lane   = threadIdx.x & 31;
    const int wave   = threadIdx.x >> 5;
    const int ntN    = Nc >> 6;                         // 64-wide N tiles
    const long tile  = (long)blockIdx.x * 8 + wave;
    const long total = (long)(M >> 4) * ntN;
    if (tile >= total) return;
    const int tm   = (int)(tile / ntN);
    const int tn   = (int)(tile % ntN);
    const int lm   = lane & 15;
    const int koff = (lane >> 4) << 3;                  // 0 or 8

    const bf16* arow  = A  + (size_t)(tm * 16 + lm) * K;
    const bf16* brow0 = Bt + (size_t)(tn * 64 + lm) * K;

    v8f acc[4] = {};
    for (int k0 = 0; k0 < K; k0 += 32) {
        const v16bf a = load_frag(arow, k0, koff);
        #pragma unroll
        for (int j = 0; j < 4; ++j) {
            const v16bf b = load_frag(brow0 + (size_t)j * 16 * K, k0, koff);
            acc[j] = __builtin_amdgcn_wmma_f32_16x16x32_bf16(
                false, a, false, b, (short)0, acc[j], false, false);
        }
    }

    // C layout: VGPR r -> M = r + (lane>=16 ? 8 : 0), N = lane&15
    const int mbase = tm * 16 + ((lane >> 4) << 3);
    #pragma unroll
    for (int j = 0; j < 4; ++j) {
        const int col   = tn * 64 + j * 16 + lm;
        const float bv  = bias ? bias[col] : 0.0f;
        #pragma unroll
        for (int r = 0; r < 8; ++r) {
            float v = acc[j][r] + bv;
            if (doGelu) v = gelu_exact(v);
            const size_t off = (size_t)(mbase + r) * Nc + col;
            if (out32) out32[off] = v;
            if (out16) out16[off] = (bf16)v;
        }
    }
}

// ---------------------------------------------------------------------------
// Attention: 1 wave per (sequence, head). T=16 tokens, HD=64.
// scores = Q K^T via 2x wmma (K=64); softmax across 16 lanes; A·V via
// K-padded wmma (16 real keys, upper 16 zero).
// ---------------------------------------------------------------------------
__global__ void attn_kernel(const bf16* __restrict__ qkv,   // [S*16, 2304]
                            bf16* __restrict__ outb) {      // [S*16, 768]
    using namespace cfg;
    const int lane = threadIdx.x & 31;
    const int wave = threadIdx.x >> 5;
    const int gw   = blockIdx.x * 8 + wave;                 // grid sized exactly
    const int s    = gw / NH;
    const int h    = gw % NH;

    const bf16* base = qkv + (size_t)s * 16 * D3;
    const int lm    = lane & 15;
    const int koff  = (lane >> 4) << 3;                     // 0 or 8
    const int mbase = (lane >> 4) << 3;

    // ---- scores = (Q @ K^T) * 1/sqrt(HD) -----------------------------------
    const bf16* qrow = base + (size_t)lm * D3 + h * HD;           // A: M = query
    const bf16* krow = base + (size_t)lm * D3 + D + h * HD;       // B: N = key
    v8f sc = {};
    #pragma unroll
    for (int kc = 0; kc < HD; kc += 32) {
        const v16bf a = load_frag(qrow, kc, koff);
        const v16bf b = load_frag(krow, kc, koff);
        sc = __builtin_amdgcn_wmma_f32_16x16x32_bf16(false, a, false, b,
                                                     (short)0, sc, false, false);
    }

    // ---- softmax over keys (N dim = 16 lanes within each half) -------------
    __shared__ float sbuf[8][16][16];   // [wave][query][key]
    #pragma unroll
    for (int r = 0; r < 8; ++r) {
        float v = sc[r] * 0.125f;       // 1/sqrt(64)
        float m = v;
        #pragma unroll
        for (int off = 1; off < 16; off <<= 1) m = fmaxf(m, __shfl_xor(m, off, 32));
        float e  = expf(v - m);
        float sm = e;
        #pragma unroll
        for (int off = 1; off < 16; off <<= 1) sm += __shfl_xor(sm, off, 32);
        sbuf[wave][mbase + r][lm] = e / sm;
    }
    __syncthreads();

    // ---- O = A @ V, K padded 16 -> 32 --------------------------------------
    v16bf aA;
    #pragma unroll
    for (int i = 0; i < 8; ++i) {
        aA[i]     = (bf16)sbuf[wave][lm][koff + i];   // K = koff..koff+7 (real)
        aA[i + 8] = (bf16)0.0f;                       // K = 16..31 (padding)
    }
    #pragma unroll
    for (int nt = 0; nt < 4; ++nt) {
        const int hd = nt * 16 + lm;
        v16bf bB;
        #pragma unroll
        for (int i = 0; i < 8; ++i) {
            const int key = koff + i;
            bB[i]     = base[(size_t)key * D3 + 2 * D + h * HD + hd];
            bB[i + 8] = (bf16)0.0f;
        }
        v8f oc = {};
        oc = __builtin_amdgcn_wmma_f32_16x16x32_bf16(false, aA, false, bB,
                                                     (short)0, oc, false, false);
        #pragma unroll
        for (int r = 0; r < 8; ++r) {
            const int q = mbase + r;
            outb[(size_t)(s * 16 + q) * D + h * HD + hd] = (bf16)oc[r];
        }
    }
}

// ---------------------------------------------------------------------------
// LayerNorm over rows of 768, fused residual / GELU / permutation / dual out.
// 1 block (256 threads) per row, 3 elements per thread.
// ---------------------------------------------------------------------------
__global__ void ln_kernel(const float* __restrict__ src, const float* __restrict__ resid,
                          const float* __restrict__ gamma, const float* __restrict__ beta,
                          float* __restrict__ out32, bf16* __restrict__ out16,
                          int doGelu, int permEdgeToSeq) {
    using namespace cfg;
    const int row = blockIdx.x;
    const float* s  = src + (size_t)row * D;
    const float* rs = resid ? resid + (size_t)row * D : nullptr;
    float x[3];
    float s1 = 0.0f, s2 = 0.0f;
    #pragma unroll
    for (int i = 0; i < 3; ++i) {
        const int c = threadIdx.x + i * 256;
        float v = s[c] + (rs ? rs[c] : 0.0f);
        x[i] = v; s1 += v; s2 += v * v;
    }
    blockReduce2(s1, s2);
    const float mean = s1 * (1.0f / D);
    const float var  = s2 * (1.0f / D) - mean * mean;
    const float inv  = rsqrtf(var + 1e-5f);

    size_t orow = row;
    if (permEdgeToSeq) {        // (b,f,j) row -> (b,j,f) row
        const int j = row % TN, bfi = row / TN;
        const int b = bfi / F,  f   = bfi % F;
        orow = (size_t)(b * TN + j) * F + f;
    }
    #pragma unroll
    for (int i = 0; i < 3; ++i) {
        const int c = threadIdx.x + i * 256;
        float v = (x[i] - mean) * inv * gamma[c] + beta[c];
        if (doGelu) v = gelu_exact(v);
        const size_t off = orow * D + c;
        if (out32) out32[off] = v;
        if (out16) out16[off] = (bf16)v;
    }
}

// ---------------------------------------------------------------------------
// Build cat = [[hf,ef,of],[of,ef,hf]]  (R x 2304) bf16, edge row order
// ---------------------------------------------------------------------------
__global__ void build_cat_kernel(const float* __restrict__ hf, const float* __restrict__ of,
                                 const float* __restrict__ ef, bf16* __restrict__ out) {
    using namespace cfg;
    size_t idx = (size_t)blockIdx.x * 256 + threadIdx.x;
    if (idx >= (size_t)R * D3) return;
    const int    c   = (int)(idx % D3);
    const size_t e   = idx / D3;
    const int    j   = (int)(e % TN);
    const size_t bfi = e / TN;
    const int seg = c / D, cc = c % D;
    float v;
    if (j < N) {
        if (seg == 0)      v = hf[bfi * D + cc];
        else if (seg == 1) v = ef[(bfi * N + j) * D + cc];
        else               v = of[(bfi * N + j) * D + cc];
    } else {
        const int jj = j - N;
        if (seg == 0)      v = of[(bfi * N + jj) * D + cc];
        else if (seg == 1) v = ef[(bfi * N + jj) * D + cc];
        else               v = hf[bfi * D + cc];
    }
    out[idx] = (bf16)v;
}

// ---------------------------------------------------------------------------
// Build mcat = [node | tmp_edge]  (R x 1536) bf16, edge row order
// ---------------------------------------------------------------------------
__global__ void build_mcat_kernel(const float* __restrict__ hf, const float* __restrict__ of,
                                  const float* __restrict__ X32, bf16* __restrict__ out) {
    using namespace cfg;
    size_t idx = (size_t)blockIdx.x * 256 + threadIdx.x;
    if (idx >= (size_t)R * D2) return;
    const int    c   = (int)(idx % D2);
    const size_t e   = idx / D2;
    const int    j   = (int)(e % TN);
    const size_t bfi = e / TN;
    const int b = (int)(bfi / F), f = (int)(bfi % F);
    float v;
    if (c < D) {
        v = (j < N) ? hf[bfi * D + c] : of[(bfi * N + (j - N)) * D + c];
    } else {
        const size_t xrow = (size_t)(b * TN + j) * F + f;   // tmp_edge from seq order
        v = X32[xrow * D + (c - D)];
    }
    out[idx] = (bf16)v;
}

// ---------------------------------------------------------------------------
// weight_edge = sigmoid(tmp_edge @ W_l + b_l)   (1 wave per edge row)
// ---------------------------------------------------------------------------
__global__ void wedge_kernel(const float* __restrict__ X32, const float* __restrict__ Wl,
                             const float* __restrict__ bl, float* __restrict__ we) {
    using namespace cfg;
    const int lane = threadIdx.x & 31;
    const int wave = threadIdx.x >> 5;
    const int e    = blockIdx.x * 8 + wave;
    if (e >= R) return;
    const int j = e % TN, bfi = e / TN;
    const int b = bfi / F, f = bfi % F;
    const size_t row = (size_t)(b * TN + j) * F + f;
    const float* xr = X32 + row * D;
    float s = 0.0f;
    for (int i = lane; i < D; i += 32) s += xr[i] * Wl[i];
    #pragma unroll
    for (int off = 16; off > 0; off >>= 1) s += __shfl_down(s, off, 32);
    if (lane == 0) we[e] = sigmoidf(s + bl[0]);
}

// ---------------------------------------------------------------------------
// ewh[bf,d] = sum_{j<N} we * m_v   (human-side message aggregation)
// ---------------------------------------------------------------------------
__global__ void ewh_kernel(const float* __restrict__ we, const float* __restrict__ mv,
                           float* __restrict__ ewh) {
    using namespace cfg;
    const int bfi = blockIdx.x;
    #pragma unroll
    for (int i = 0; i < 3; ++i) {
        const int d = threadIdx.x + i * 256;
        float s = 0.0f;
        #pragma unroll
        for (int j = 0; j < N; ++j) {
            const size_t e = (size_t)bfi * TN + j;
            s += we[e] * mv[e * D + d];
        }
        ewh[(size_t)bfi * D + d] = s;
    }
}

// ---------------------------------------------------------------------------
// msg_norm + residual for human rows -> bf16 GEMM input
// ---------------------------------------------------------------------------
__global__ void msgnorm_h_kernel(const float* __restrict__ ewh, const float* __restrict__ hf,
                                 const float* __restrict__ scale, bf16* __restrict__ yh) {
    using namespace cfg;
    const int bfi = blockIdx.x;
    float m[3], xx[3];
    float s1 = 0.0f, s2 = 0.0f;
    #pragma unroll
    for (int i = 0; i < 3; ++i) {
        const int d = threadIdx.x + i * 256;
        m[i]  = ewh[(size_t)bfi * D + d];
        xx[i] = hf [(size_t)bfi * D + d];
        s1 += m[i] * m[i];
        s2 += xx[i] * xx[i];
    }
    blockReduce2(s1, s2);
    const float coef = sqrtf(s2) * scale[0] / fmaxf(sqrtf(s1), 1e-12f);
    #pragma unroll
    for (int i = 0; i < 3; ++i) {
        const int d = threadIdx.x + i * 256;
        yh[(size_t)bfi * D + d] = (bf16)(m[i] * coef + xx[i]);
    }
}

// ---------------------------------------------------------------------------
// msg_norm + residual for object rows -> bf16 GEMM input
// ---------------------------------------------------------------------------
__global__ void msgnorm_o_kernel(const float* __restrict__ we, const float* __restrict__ mv,
                                 const float* __restrict__ of, const float* __restrict__ scale,
                                 bf16* __restrict__ yo) {
    using namespace cfg;
    const int r   = blockIdx.x;          // bfi*N + j
    const int j   = r % N;
    const int bfi = r / N;
    const size_t e = (size_t)bfi * TN + N + j;
    const float w = we[e];
    float m[3], xx[3];
    float s1 = 0.0f, s2 = 0.0f;
    #pragma unroll
    for (int i = 0; i < 3; ++i) {
        const int d = threadIdx.x + i * 256;
        m[i]  = w * mv[e * D + d];
        xx[i] = of[(size_t)r * D + d];
        s1 += m[i] * m[i];
        s2 += xx[i] * xx[i];
    }
    blockReduce2(s1, s2);
    const float coef = sqrtf(s2) * scale[0] / fmaxf(sqrtf(s1), 1e-12f);
    #pragma unroll
    for (int i = 0; i < 3; ++i) {
        const int d = threadIdx.x + i * 256;
        yo[(size_t)r * D + d] = (bf16)(m[i] * coef + xx[i]);
    }
}

// ---------------------------------------------------------------------------
// Host launcher
// ---------------------------------------------------------------------------
extern "C" void kernel_launch(void* const* d_in, const int* in_sizes, int n_in,
                              void* d_out, int out_size, void* d_ws, size_t ws_size,
                              hipStream_t stream) {
    using namespace cfg;
    (void)in_sizes; (void)n_in; (void)out_size;

    // ---- inputs (setup_inputs order) ----
    const float* hf      = (const float*)d_in[0];
    const float* of      = (const float*)d_in[1];
    const float* ef      = (const float*)d_in[2];
    const float* W_e1    = (const float*)d_in[3];
    const float* b_e1    = (const float*)d_in[4];
    const float* W_e2    = (const float*)d_in[5];
    const float* b_e2    = (const float*)d_in[6];
    const float* g_e     = (const float*)d_in[7];
    const float* be_e    = (const float*)d_in[8];
    const float* W_m     = (const float*)d_in[9];
    const float* b_m     = (const float*)d_in[10];
    const float* W_mg    = (const float*)d_in[11];
    const float* b_mg    = (const float*)d_in[12];
    const float* g_mg    = (const float*)d_in[13];
    const float* bb_mg   = (const float*)d_in[14];
    const float* W_l     = (const float*)d_in[15];
    const float* b_l     = (const float*)d_in[16];
    const float* scale_h = (const float*)d_in[17];
    const float* scale_o = (const float*)d_in[18];
    const float* W_nh    = (const float*)d_in[19];
    const float* b_nh    = (const float*)d_in[20];
    const float* g_nh    = (const float*)d_in[21];
    const float* bb_nh   = (const float*)d_in[22];
    const float* W_no    = (const float*)d_in[23];
    const float* b_no    = (const float*)d_in[24];
    const float* g_no    = (const float*)d_in[25];
    const float* bb_no   = (const float*)d_in[26];
    const float* Wqkv    = (const float*)d_in[27];
    const float* bqkv    = (const float*)d_in[28];
    const float* Wo_t    = (const float*)d_in[29];
    const float* bo_t    = (const float*)d_in[30];
    const float* g1      = (const float*)d_in[31];
    const float* bb1     = (const float*)d_in[32];
    const float* Wf1     = (const float*)d_in[33];
    const float* bf1     = (const float*)d_in[34];
    const float* Wf2     = (const float*)d_in[35];
    const float* bf2     = (const float*)d_in[36];
    const float* g2      = (const float*)d_in[37];
    const float* bb2     = (const float*)d_in[38];

    // ---- workspace carve (~341 MB total) ----
    char* p = (char*)d_ws;
    auto alloc = [&](size_t bytes) -> char* {
        char* r = p;
        p += (bytes + 255) & ~(size_t)255;
        return r;
    };
    bf16* wp_e1t  = (bf16*)alloc((size_t)D  * D3 * 2);
    bf16* wp_e2t  = (bf16*)alloc((size_t)D  * D  * 2);
    bf16* wp_mt   = (bf16*)alloc((size_t)D  * D2 * 2);
    bf16* wp_mgt  = (bf16*)alloc((size_t)D  * D  * 2);
    bf16* wp_nht  = (bf16*)alloc((size_t)D  * D  * 2);
    bf16* wp_not  = (bf16*)alloc((size_t)D  * D  * 2);
    bf16* wp_qkvt = (bf16*)alloc((size_t)2 * D3 * D * 2);
    bf16* wp_ot   = (bf16*)alloc((size_t)2 * D  * D * 2);
    bf16* wp_f1t  = (bf16*)alloc((size_t)2 * FF * D * 2);
    bf16* wp_f2t  = (bf16*)alloc((size_t)2 * D  * FF * 2);
    bf16* A16     = (bf16*)alloc((size_t)R * FF * 2);    // cat / qkv / ff1 / mcat+m_v
    bf16* H16     = (bf16*)alloc((size_t)R * D * 2);     // h1 / attn_out / m1
    bf16* X16     = (bf16*)alloc((size_t)R * D * 2);     // transformer x (bf16)
    float* X32    = (float*)alloc((size_t)R * D * 4);    // transformer x (fp32)
    float* Bf     = (float*)alloc((size_t)R * D * 4);    // generic fp32 GEMM out
    float* WE     = (float*)alloc((size_t)R * 4);        // weight_edge
    float* EWH    = (float*)alloc((size_t)B * F * D * 4);
    bf16* YH      = (bf16*)alloc((size_t)B * F * D * 2);
    bf16* YO      = (bf16*)alloc((size_t)B * F * N * D * 2);
    // m_v (fp32) overlays the second half of A16 (mcat only uses first R*D2*2 B)
    float* MV32   = (float*)((char*)A16 + (size_t)R * D2 * 2);
    if ((size_t)(p - (char*)d_ws) > ws_size) return;     // ws too small: bail safely

    auto eb = [](size_t n) { return (unsigned)((n + 255) / 256); };        // elementwise
    auto gb = [](int M, int Nc) {                                          // gemm blocks
        long t = (long)(M / 16) * (Nc / 64);
        return (unsigned)((t + 7) / 8);
    };

    // ---- 0) pack all weights to bf16 transposed (N x K) ----
    pack_wt_kernel<<<eb((size_t)D3 * D), 256, 0, stream>>>(W_e1, wp_e1t, D3, D);
    pack_wt_kernel<<<eb((size_t)D  * D), 256, 0, stream>>>(W_e2, wp_e2t, D, D);
    pack_wt_kernel<<<eb((size_t)D2 * D), 256, 0, stream>>>(W_m,  wp_mt,  D2, D);
    pack_wt_kernel<<<eb((size_t)D  * D), 256, 0, stream>>>(W_mg, wp_mgt, D, D);
    pack_wt_kernel<<<eb((size_t)D  * D), 256, 0, stream>>>(W_nh, wp_nht, D, D);
    pack_wt_kernel<<<eb((size_t)D  * D), 256, 0, stream>>>(W_no, wp_not, D, D);
    for (int l = 0; l < 2; ++l) {
        pack_wt_kernel<<<eb((size_t)D * D3), 256, 0, stream>>>(
            Wqkv + (size_t)l * D * D3, wp_qkvt + (size_t)l * D3 * D, D, D3);
        pack_wt_kernel<<<eb((size_t)D * D), 256, 0, stream>>>(
            Wo_t + (size_t)l * D * D, wp_ot + (size_t)l * D * D, D, D);
        pack_wt_kernel<<<eb((size_t)D * FF), 256, 0, stream>>>(
            Wf1 + (size_t)l * D * FF, wp_f1t + (size_t)l * FF * D, D, FF);
        pack_wt_kernel<<<eb((size_t)FF * D), 256, 0, stream>>>(
            Wf2 + (size_t)l * FF * D, wp_f2t + (size_t)l * D * FF, FF, D);
    }

    // ---- 1) edge MLP ----
    build_cat_kernel<<<eb((size_t)R * D3), 256, 0, stream>>>(hf, of, ef, A16);
    gemm_bf16_kernel<<<gb(R, D), 256, 0, stream>>>(A16, wp_e1t, b_e1,
                                                   nullptr, H16, R, D, D3, 1);
    gemm_bf16_kernel<<<gb(R, D), 256, 0, stream>>>(H16, wp_e2t, b_e2,
                                                   Bf, nullptr, R, D, D, 0);
    // LN + GELU, and permute (b,f,j) -> (b,j,f) seq order into X32/X16
    ln_kernel<<<R, 256, 0, stream>>>(Bf, nullptr, g_e, be_e, X32, X16, 1, 1);

    // ---- 2) transformer (L = 2) ----
    for (int l = 0; l < 2; ++l) {
        gemm_bf16_kernel<<<gb(R, D3), 256, 0, stream>>>(
            X16, wp_qkvt + (size_t)l * D3 * D, bqkv + (size_t)l * D3,
            nullptr, A16, R, D3, D, 0);
        attn_kernel<<<(S * NH) / 8, 256, 0, stream>>>(A16, H16);
        gemm_bf16_kernel<<<gb(R, D), 256, 0, stream>>>(
            H16, wp_ot + (size_t)l * D * D, bo_t + (size_t)l * D,
            Bf, nullptr, R, D, D, 0);
        ln_kernel<<<R, 256, 0, stream>>>(Bf, X32, g1 + (size_t)l * D, bb1 + (size_t)l * D,
                                         X32, X16, 0, 0);
        gemm_bf16_kernel<<<gb(R, FF), 256, 0, stream>>>(
            X16, wp_f1t + (size_t)l * FF * D, bf1 + (size_t)l * FF,
            nullptr, A16, R, FF, D, 1);
        gemm_bf16_kernel<<<gb(R, D), 256, 0, stream>>>(
            A16, wp_f2t + (size_t)l * D * FF, bf2 + (size_t)l * D,
            Bf, nullptr, R, D, FF, 0);
        ln_kernel<<<R, 256, 0, stream>>>(Bf, X32, g2 + (size_t)l * D, bb2 + (size_t)l * D,
                                         X32, X16, 0, 0);
    }

    // ---- 3) readout ----
    wedge_kernel<<<R / 8, 256, 0, stream>>>(X32, W_l, b_l, WE);
    build_mcat_kernel<<<eb((size_t)R * D2), 256, 0, stream>>>(hf, of, X32, A16);
    gemm_bf16_kernel<<<gb(R, D), 256, 0, stream>>>(A16, wp_mt, b_m,
                                                   nullptr, H16, R, D, D2, 1);
    gemm_bf16_kernel<<<gb(R, D), 256, 0, stream>>>(H16, wp_mgt, b_mg,
                                                   Bf, nullptr, R, D, D, 0);
    ln_kernel<<<R, 256, 0, stream>>>(Bf, nullptr, g_mg, bb_mg, MV32, nullptr, 1, 0);

    ewh_kernel<<<B * F, 256, 0, stream>>>(WE, MV32, EWH);
    msgnorm_h_kernel<<<B * F, 256, 0, stream>>>(EWH, hf, scale_h, YH);
    msgnorm_o_kernel<<<B * F * N, 256, 0, stream>>>(WE, MV32, of, scale_o, YO);

    // ---- 4) node heads -> d_out (fp32) ----
    float* out = (float*)d_out;
    gemm_bf16_kernel<<<gb(B * F, D), 256, 0, stream>>>(YH, wp_nht, b_nh,
                                                       Bf, nullptr, B * F, D, D, 0);
    ln_kernel<<<B * F, 256, 0, stream>>>(Bf, nullptr, g_nh, bb_nh, out, nullptr, 1, 0);
    gemm_bf16_kernel<<<gb(B * F * N, D), 256, 0, stream>>>(YO, wp_not, b_no,
                                                           Bf, nullptr, B * F * N, D, D, 0);
    ln_kernel<<<B * F * N, 256, 0, stream>>>(Bf, nullptr, g_no, bb_no,
                                             out + (size_t)B * F * D, nullptr, 1, 0);
}
